// Model5_9620726743221
// MI455X (gfx1250) — matrix operand order repaired and checked by reading, not compiled, not run on hardware
//
#include <hip/hip_runtime.h>
#include <hip/hip_bf16.h>
#include <math.h>

typedef __attribute__((ext_vector_type(16))) _Float16 v16h;
typedef __attribute__((ext_vector_type(8)))  float    v8f;

#define WMMA_F16(a, b, c) \
  __builtin_amdgcn_wmma_f32_16x16x32_f16(false, (a), false, (b), (short)0, (c), false, false)

// ---------------- helpers ----------------

// order-preserving float -> uint map (for atomicMax on floats incl. negatives)
__device__ static inline unsigned fmapu(float f) {
  unsigned u = __float_as_uint(f);
  return (u & 0x80000000u) ? ~u : (u | 0x80000000u);
}
__device__ static inline float funmap(unsigned u) {
  return (u & 0x80000000u) ? __uint_as_float(u ^ 0x80000000u) : __uint_as_float(~u);
}

// A fragment (16x32 f16 tile) from per-wave LDS feat[16][KP], k-block kb.
// ISA layout: lanes 0-15 = M rows, VGPR v<4 -> K=2v(+1)+8*half, v>=4 -> K=16+2(v-4)(+1)+8*half
__device__ static inline v16h load_a_frag(const _Float16* feat, int KP, int kb, int lane) {
  const int m = lane & 15, half = lane >> 4;
  v16h a;
#pragma unroll
  for (int v = 0; v < 8; ++v) {
    int k = kb + ((v < 4) ? (2 * v) : (16 + 2 * (v - 4))) + 8 * half;
    a[2 * v]     = feat[m * KP + k];
    a[2 * v + 1] = feat[m * KP + k + 1];
  }
  return a;
}

// B fragment (32x16 f16 tile) from global f32 weights W[K][NOUT] row-major.
// ISA layout: col = lane%16; VGPR v<4 -> K=2v(+1)+8*half, v>=4 -> K=16+2(v-4)(+1)+8*half (matches A).
__device__ static inline v16h load_b_frag(const float* W, int K, int NOUT, int kb, int tile, int lane) {
  const int n = (lane & 15) + 16 * tile, half = lane >> 4;
  v16h b;
#pragma unroll
  for (int v = 0; v < 8; ++v) {
    int k = kb + ((v < 4) ? (2 * v) : (16 + 2 * (v - 4))) + 8 * half;
    float b0 = (k     < K && n < NOUT) ? W[k * NOUT + n]       : 0.f;
    float b1 = (k + 1 < K && n < NOUT) ? W[(k + 1) * NOUT + n] : 0.f;
    b[2 * v]     = (_Float16)b0;
    b[2 * v + 1] = (_Float16)b1;
  }
  return b;
}

// ---------------- simple kernels ----------------

__global__ void k_zero(float* __restrict__ p, int n) {
  int i = blockIdx.x * blockDim.x + threadIdx.x;
  if (i < n) p[i] = 0.f;
}

__global__ void k_gat_init(float* __restrict__ acc, unsigned* __restrict__ mU,
                           float* __restrict__ sumw, int N) {
  int i = blockIdx.x * blockDim.x + threadIdx.x;
  if (i < N * 10) acc[i] = 0.f;
  if (i < N) { mU[i] = fmapu(-INFINITY); sumw[i] = 0.f; }
}

// xl = xcat @ wl, xr = xcat @ wr ; xcat = x1 (layer1) or [h, x1] (layers 2/3)
__global__ void k_xform(const float* __restrict__ x1, const float* __restrict__ h,
                        int use_h, int Fin,
                        const float* __restrict__ wl, const float* __restrict__ wr,
                        float* __restrict__ xl, float* __restrict__ xr, int N) {
  int n = blockIdx.x * blockDim.x + threadIdx.x;
  if (n >= N) return;
  float xc[25];
  if (use_h) {
#pragma unroll
    for (int k = 0; k < 10; ++k) xc[k] = h[n * 10 + k];
#pragma unroll
    for (int k = 0; k < 15; ++k) xc[10 + k] = x1[n * 15 + k];
  } else {
#pragma unroll
    for (int k = 0; k < 15; ++k) xc[k] = x1[n * 15 + k];
  }
#pragma unroll
  for (int j = 0; j < 10; ++j) {
    float al = 0.f, ar = 0.f;
    for (int k = 0; k < Fin; ++k) { al += xc[k] * wl[k * 10 + j]; ar += xc[k] * wr[k * 10 + j]; }
    xl[n * 10 + j] = al;
    xr[n * 10 + j] = ar;
  }
}

// e_i = leaky_relu(xl[s]+xr[d]) . att ; segment-max into mU[d]
__global__ void k_edge_logits(const int* __restrict__ src, const int* __restrict__ dst,
                              const float* __restrict__ xl, const float* __restrict__ xr,
                              const float* __restrict__ att,
                              float* __restrict__ e, unsigned* __restrict__ mU, int E, int N) {
  int i = blockIdx.x * blockDim.x + threadIdx.x;
  if (i >= E + N) return;
  if (i + 16384 < E) { __builtin_prefetch(src + i + 16384, 0, 0); __builtin_prefetch(dst + i + 16384, 0, 0); }
  int s = (i < E) ? src[i] : (i - E);
  int d = (i < E) ? dst[i] : s;
  float acc = 0.f;
#pragma unroll
  for (int j = 0; j < 10; ++j) {
    float t = xl[s * 10 + j] + xr[d * 10 + j];
    t = (t > 0.f) ? t : 0.2f * t;
    acc += t * att[j];
  }
  e[i] = acc;
  atomicMax(&mU[d], fmapu(acc));
}

// w_i = exp(e_i - m[d]) ; segment-sum into sumw[d]
__global__ void k_edge_exp(const int* __restrict__ dst, const float* __restrict__ e,
                           const unsigned* __restrict__ mU, float* __restrict__ w,
                           float* __restrict__ sumw, int E, int N) {
  int i = blockIdx.x * blockDim.x + threadIdx.x;
  if (i >= E + N) return;
  int d = (i < E) ? dst[i] : (i - E);
  float wv = __expf(e[i] - funmap(mU[d]));
  w[i] = wv;
  atomicAdd(&sumw[d], wv);
}

// acc[d] += (w_i / sumw[d]) * xl[s]
__global__ void k_edge_scatter(const int* __restrict__ src, const int* __restrict__ dst,
                               const float* __restrict__ w, const float* __restrict__ sumw,
                               const float* __restrict__ xl, float* __restrict__ acc, int E, int N) {
  int i = blockIdx.x * blockDim.x + threadIdx.x;
  if (i >= E + N) return;
  if (i + 16384 < E) { __builtin_prefetch(w + i + 16384, 0, 0); }
  int s = (i < E) ? src[i] : (i - E);
  int d = (i < E) ? dst[i] : s;
  float alpha = w[i] / sumw[d];
#pragma unroll
  for (int j = 0; j < 10; ++j) atomicAdd(&acc[d * 10 + j], alpha * xl[s * 10 + j]);
}

// h = relu(acc + b)
__global__ void k_finalize(const float* __restrict__ acc, const float* __restrict__ b,
                           float* __restrict__ h, int N) {
  int i = blockIdx.x * blockDim.x + threadIdx.x;
  if (i < N * 10) h[i] = fmaxf(acc[i] + b[i % 10], 0.f);
}

// ---------------- WMMA MLP kernels (one wave = 16 rows) ----------------

// attack: feat 48 -> relu(.@aaa_w+aaa_b) [20] -> .@bbb_w+bbb_b -> atomicAdd p[move]
__global__ void __launch_bounds__(256) k_attack(
    const float* __restrict__ h, const float* __restrict__ x1,
    const int* __restrict__ asrc, const int* __restrict__ adst,
    const float* __restrict__ aarm, const int* __restrict__ amove,
    const float* __restrict__ aaaW, const float* __restrict__ aaaB,
    const float* __restrict__ bbbW, const float* __restrict__ bbbB,
    float* __restrict__ p, int Ma) {
  __shared__ _Float16 featS[8][16 * 64];
  __shared__ float    hidS[8][16 * 32];
  const int lane = threadIdx.x & 31, wid = threadIdx.x >> 5;
  _Float16* feat = featS[wid];
  float*    hid  = hidS[wid];
  const int rl = lane & 15, csel = lane >> 4;

  const v16h b00 = load_b_frag(aaaW, 48, 20, 0,  0, lane);
  const v16h b10 = load_b_frag(aaaW, 48, 20, 32, 0, lane);
  const v16h b01 = load_b_frag(aaaW, 48, 20, 0,  1, lane);
  const v16h b11 = load_b_frag(aaaW, 48, 20, 32, 1, lane);
  const float bs0 = aaaB[rl];
  const float bs1 = (rl + 16 < 20) ? aaaB[rl + 16] : 0.f;

  for (int bb = blockIdx.x * 128; bb < Ma; bb += gridDim.x * 128) {
    const int row = bb + wid * 16 + rl;
    const bool valid = row < Ma;
    const int rs = valid ? asrc[row] : 0;
    const int rd = valid ? adst[row] : 0;
    const float ar = valid ? aarm[row] : 0.f;
    const float xd3 = valid ? x1[rd * 15 + 3] : 0.f;
    const float xd4 = valid ? x1[rd * 15 + 4] : 0.f;
    for (int c = 32 * csel; c < 32 * csel + 32; ++c) {
      float fv = 0.f;
      if (valid) {
        if      (c < 10) fv = h[rs * 10 + c];
        else if (c < 20) fv = h[rd * 10 + (c - 10)];
        else if (c < 32) fv = x1[rs * 15 + 3 + (c - 20)];
        else if (c < 46) fv = x1[rd * 15 + 1 + (c - 32)];
        else if (c == 46) fv = ar;
        else if (c == 47) fv = 0.6f * ar - 0.7f * (xd3 + xd4);
      }
      feat[rl * 64 + c] = (_Float16)fv;
    }
    __syncthreads();
    v16h a0 = load_a_frag(feat, 64, 0,  lane);
    v16h a1 = load_a_frag(feat, 64, 32, lane);
    v8f c0 = {}; c0 = WMMA_F16(a0, b00, c0); c0 = WMMA_F16(a1, b10, c0);
    v8f c1 = {}; c1 = WMMA_F16(a0, b01, c1); c1 = WMMA_F16(a1, b11, c1);
#pragma unroll
    for (int v = 0; v < 8; ++v) {
      int m = 8 * csel + v;
      hid[m * 32 + rl]      = fmaxf(c0[v] + bs0, 0.f);
      hid[m * 32 + rl + 16] = (rl + 16 < 20) ? fmaxf(c1[v] + bs1, 0.f) : 0.f;
    }
    __syncthreads();
    if (lane < 16) {
      const int row2 = bb + wid * 16 + lane;
      if (row2 < Ma) {
        float s = bbbB[0];
#pragma unroll
        for (int c = 0; c < 20; ++c) s += hid[lane * 32 + c] * bbbW[c];
        atomicAdd(&p[amove[row2]], s);
      }
    }
    __syncthreads();
  }
}

// deploy: feat 23 -> relu(.@ccc_w+ccc_b) [20] -> .@ddd_w+ddd_b -> atomicAdd p[move]
__global__ void __launch_bounds__(256) k_deploy(
    const float* __restrict__ h, const float* __restrict__ x1,
    const int* __restrict__ tgt, const float* __restrict__ darm, const int* __restrict__ dmove,
    const float* __restrict__ cccW, const float* __restrict__ cccB,
    const float* __restrict__ dddW, const float* __restrict__ dddB,
    float* __restrict__ p, int Md) {
  __shared__ _Float16 featS[8][16 * 32];
  __shared__ float    hidS[8][16 * 32];
  const int lane = threadIdx.x & 31, wid = threadIdx.x >> 5;
  _Float16* feat = featS[wid];
  float*    hid  = hidS[wid];
  const int rl = lane & 15, csel = lane >> 4;

  const v16h b0 = load_b_frag(cccW, 23, 20, 0, 0, lane);
  const v16h b1 = load_b_frag(cccW, 23, 20, 0, 1, lane);
  const float bs0 = cccB[rl];
  const float bs1 = (rl + 16 < 20) ? cccB[rl + 16] : 0.f;

  for (int bb = blockIdx.x * 128; bb < Md; bb += gridDim.x * 128) {
    const int row = bb + wid * 16 + rl;
    const bool valid = row < Md;
    const int rt = valid ? tgt[row] : 0;
    const float ar = valid ? darm[row] : 0.f;
    for (int c = 16 * csel; c < 16 * csel + 16; ++c) {
      float fv = 0.f;
      if (valid) {
        if      (c < 10) fv = h[rt * 10 + c];
        else if (c < 22) fv = x1[rt * 15 + 3 + (c - 10)];
        else if (c == 22) fv = ar;
      }
      feat[rl * 32 + c] = (_Float16)fv;
    }
    __syncthreads();
    v16h a = load_a_frag(feat, 32, 0, lane);
    v8f c0 = {}; c0 = WMMA_F16(a, b0, c0);
    v8f c1 = {}; c1 = WMMA_F16(a, b1, c1);
#pragma unroll
    for (int v = 0; v < 8; ++v) {
      int m = 8 * csel + v;
      hid[m * 32 + rl]      = fmaxf(c0[v] + bs0, 0.f);
      hid[m * 32 + rl + 16] = (rl + 16 < 20) ? fmaxf(c1[v] + bs1, 0.f) : 0.f;
    }
    __syncthreads();
    if (lane < 16) {
      const int row2 = bb + wid * 16 + lane;
      if (row2 < Md) {
        float s = dddB[0];
#pragma unroll
        for (int c = 0; c < 20; ++c) s += hid[lane * 32 + c] * dddW[c];
        atomicAdd(&p[dmove[row2]], s);
      }
    }
    __syncthreads();
  }
}

// value head: feat 29 -> relu(.@lin_w+lin_b) [15] -> .@lin2_w+lin2_b -> atomicAdd vsum
__global__ void __launch_bounds__(256) k_value(
    const float* __restrict__ h, const float* __restrict__ x1, const float* __restrict__ x2,
    const float* __restrict__ linW, const float* __restrict__ linB,
    const float* __restrict__ lin2W, const float* __restrict__ lin2B,
    float* __restrict__ vsum, int N) {
  __shared__ _Float16 featS[8][16 * 32];
  __shared__ float    hidS[8][16 * 16];
  const int lane = threadIdx.x & 31, wid = threadIdx.x >> 5;
  _Float16* feat = featS[wid];
  float*    hid  = hidS[wid];
  const int rl = lane & 15, csel = lane >> 4;

  const v16h b0 = load_b_frag(linW, 29, 15, 0, 0, lane);
  const float bs0 = (rl < 15) ? linB[rl] : 0.f;

  for (int bb = blockIdx.x * 128; bb < N; bb += gridDim.x * 128) {
    const int row = bb + wid * 16 + rl;
    const bool valid = row < N;
    for (int c = 16 * csel; c < 16 * csel + 16; ++c) {
      float fv = 0.f;
      if (valid) {
        if      (c < 10) fv = h[row * 10 + c];
        else if (c < 25) fv = x1[row * 15 + (c - 10)];
        else if (c < 29) fv = x2[c - 25];
      }
      feat[rl * 32 + c] = (_Float16)fv;
    }
    __syncthreads();
    v16h a = load_a_frag(feat, 32, 0, lane);
    v8f c0 = {}; c0 = WMMA_F16(a, b0, c0);
#pragma unroll
    for (int v = 0; v < 8; ++v) {
      int m = 8 * csel + v;
      hid[m * 16 + rl] = (rl < 15) ? fmaxf(c0[v] + bs0, 0.f) : 0.f;
    }
    __syncthreads();
    if (lane < 16) {
      const int row2 = bb + wid * 16 + lane;
      if (row2 < N) {
        float vr = lin2B[0];
#pragma unroll
        for (int c = 0; c < 15; ++c) vr += hid[lane * 16 + c] * lin2W[c];
        atomicAdd(vsum, vr);
      }
    }
    __syncthreads();
  }
}

// log_softmax(p) -> out[1..M] ; V = tanh(vsum/N) -> out[0]
__global__ void k_final(const float* __restrict__ p, const float* __restrict__ vsum,
                        float* __restrict__ out, int M, int Nn) {
  __shared__ float red[1024];
  const int t = threadIdx.x;
  float mx = -INFINITY;
  for (int i = t; i < M; i += 1024) mx = fmaxf(mx, p[i]);
  red[t] = mx; __syncthreads();
  for (int s = 512; s > 0; s >>= 1) { if (t < s) red[t] = fmaxf(red[t], red[t + s]); __syncthreads(); }
  mx = red[0]; __syncthreads();
  float se = 0.f;
  for (int i = t; i < M; i += 1024) se += expf(p[i] - mx);
  red[t] = se; __syncthreads();
  for (int s = 512; s > 0; s >>= 1) { if (t < s) red[t] += red[t + s]; __syncthreads(); }
  float lse = logf(red[0]) + mx;
  for (int i = t; i < M; i += 1024) out[1 + i] = p[i] - lse;
  if (t == 0) out[0] = tanhf(vsum[0] / (float)Nn);
}

// ---------------- launch ----------------

extern "C" void kernel_launch(void* const* d_in, const int* in_sizes, int n_in,
                              void* d_out, int out_size, void* d_ws, size_t ws_size,
                              hipStream_t stream) {
  (void)n_in; (void)ws_size;
  const float* x1   = (const float*)d_in[0];
  const float* x2   = (const float*)d_in[1];
  const int*   edges= (const int*)d_in[2];
  const int*   asrc = (const int*)d_in[3];
  const int*   adst = (const int*)d_in[4];
  const float* aarm = (const float*)d_in[5];
  const int*   amove= (const int*)d_in[6];
  const int*   dtgt = (const int*)d_in[7];
  const float* darm = (const float*)d_in[8];
  const int*   dmove= (const int*)d_in[9];
  const float* g_wl[3]  = {(const float*)d_in[11], (const float*)d_in[15], (const float*)d_in[19]};
  const float* g_wr[3]  = {(const float*)d_in[12], (const float*)d_in[16], (const float*)d_in[20]};
  const float* g_att[3] = {(const float*)d_in[13], (const float*)d_in[17], (const float*)d_in[21]};
  const float* g_b[3]   = {(const float*)d_in[14], (const float*)d_in[18], (const float*)d_in[22]};
  const float* linW  = (const float*)d_in[23];
  const float* linB  = (const float*)d_in[24];
  const float* lin2W = (const float*)d_in[25];
  const float* lin2B = (const float*)d_in[26];
  const float* aaaW  = (const float*)d_in[27];
  const float* aaaB  = (const float*)d_in[28];
  const float* bbbW  = (const float*)d_in[29];
  const float* bbbB  = (const float*)d_in[30];
  const float* cccW  = (const float*)d_in[31];
  const float* cccB  = (const float*)d_in[32];
  const float* dddW  = (const float*)d_in[33];
  const float* dddB  = (const float*)d_in[34];

  const int N  = in_sizes[0] / 15;
  const int E  = in_sizes[2] / 2;
  const int Ma = in_sizes[3];
  const int Md = in_sizes[7];
  const int M  = out_size - 1;   // num_moves
  const int T  = E + N;          // edges + self loops
  const int* esrc = edges;
  const int* edst = edges + E;

  char* w8 = (char*)d_ws;
  size_t off = 0;
  auto carve = [&](size_t bytes) -> void* {
    void* r = (void*)(w8 + off);
    off += (bytes + 255) & ~(size_t)255;
    return r;
  };
  float*    h    = (float*)carve((size_t)N * 10 * 4);
  float*    xl   = (float*)carve((size_t)N * 10 * 4);
  float*    xr   = (float*)carve((size_t)N * 10 * 4);
  float*    acc  = (float*)carve((size_t)N * 10 * 4);
  unsigned* mU   = (unsigned*)carve((size_t)N * 4);
  float*    sumw = (float*)carve((size_t)N * 4);
  float*    ebuf = (float*)carve((size_t)T * 4);
  float*    wbuf = (float*)carve((size_t)T * 4);
  float*    p    = (float*)carve((size_t)(M + 1) * 4);
  float*    vsum = p + M;

  const int B = 256;
  k_zero<<<(M + 1 + B - 1) / B, B, 0, stream>>>(p, M + 1);

  const int FinL[3] = {15, 25, 25};
  for (int l = 0; l < 3; ++l) {
    k_gat_init<<<(N * 10 + B - 1) / B, B, 0, stream>>>(acc, mU, sumw, N);
    k_xform<<<(N + B - 1) / B, B, 0, stream>>>(x1, h, l > 0 ? 1 : 0, FinL[l],
                                               g_wl[l], g_wr[l], xl, xr, N);
    k_edge_logits<<<(T + B - 1) / B, B, 0, stream>>>(esrc, edst, xl, xr, g_att[l], ebuf, mU, E, N);
    k_edge_exp<<<(T + B - 1) / B, B, 0, stream>>>(edst, ebuf, mU, wbuf, sumw, E, N);
    k_edge_scatter<<<(T + B - 1) / B, B, 0, stream>>>(esrc, edst, wbuf, sumw, xl, acc, E, N);
    k_finalize<<<(N * 10 + B - 1) / B, B, 0, stream>>>(acc, g_b[l], h, N);
  }

  int gA = (Ma + 127) / 128; if (gA > 2048) gA = 2048;
  int gD = (Md + 127) / 128; if (gD > 2048) gD = 2048;
  int gV = (N  + 127) / 128; if (gV > 2048) gV = 2048;
  k_attack<<<gA, 256, 0, stream>>>(h, x1, asrc, adst, aarm, amove, aaaW, aaaB, bbbW, bbbB, p, Ma);
  k_deploy<<<gD, 256, 0, stream>>>(h, x1, dtgt, darm, dmove, cccW, cccB, dddW, dddB, p, Md);
  k_value <<<gV, 256, 0, stream>>>(h, x1, x2, linW, linB, lin2W, lin2B, vsum, N);
  k_final <<<1, 1024, 0, stream>>>(p, vsum, (float*)d_out, M, N);
}
